// Decoder_86397562126931
// MI455X (gfx1250) — matrix operand order repaired and checked by reading, not compiled, not run on hardware
//
#include <hip/hip_runtime.h>
#include <hip/hip_bf16.h>

// ---------------------------------------------------------------------------
// MI455X (gfx1250) decoder forward.
//   B=16, TD=TE=128, U=1024, V=32000
// - Persistent grid-synced GRU scan in f32 WMMA (16x16x4, K split 8 ways to
//   shorten the dependent accumulation chains), recurrent weights L2-resident.
// - Attention with Wa folded into the encoder side (f32 WMMA).
// - Dominant 134-GFLOP logits GEMM in bf16 WMMA (16x16x32, f32 accumulate),
//   fcW pre-packed into a contiguous-k (k = 16*half + j) B-fragment lane layout; logits emitted
//   with non-temporal stores so the 262MB write-once output does not evict
//   the 65MB packed fcW from the 192MB L2 between M-group re-reads.
// ---------------------------------------------------------------------------

#define B_   16
#define TD_  128
#define TE_  128
#define U_   1024
#define V_   32000
#define NEG_ (-1e9f)

typedef __attribute__((ext_vector_type(2)))  float  v2f;
typedef __attribute__((ext_vector_type(8)))  float  v8f;
typedef __attribute__((ext_vector_type(8)))  __bf16 v8bf;
typedef __attribute__((ext_vector_type(16))) __bf16 v16bf;

__device__ __forceinline__ v8f wmma_f32(v2f a, v2f b, v8f c) {
    return __builtin_amdgcn_wmma_f32_16x16x4_f32(false, a, false, b,
                                                 (short)0, c, false, false);
}
__device__ __forceinline__ v8f wmma_bf16(v16bf a, v16bf b, v8f c) {
    return __builtin_amdgcn_wmma_f32_16x16x32_bf16(false, a, false, b,
                                                   (short)0, c, false, false);
}

__device__ __forceinline__ float sigm(float x) { return 1.0f / (1.0f + __expf(-x)); }

// ---------------------------------------------------------------------------
// Grid barrier (generation counter + two alternating arrival counters).
// The 64 scan blocks are trivially co-resident on MI455X.
// ---------------------------------------------------------------------------
__device__ __forceinline__ void gsync(unsigned* bar, unsigned* cnt,
                                      unsigned gen, int nblocks) {
    __syncthreads();
    if (threadIdx.x == 0) {
        __threadfence();
        unsigned idx = gen & 1u;
        if (atomicAdd(&cnt[idx], 1u) == (unsigned)(nblocks - 1)) {
            cnt[idx] = 0u;
            __threadfence();
            atomicAdd(bar, 1u);
        } else {
            while (((volatile unsigned*)bar)[0] <= gen) { /* spin */ }
        }
    }
    __threadfence();
    __syncthreads();
}

// ---------------------------------------------------------------------------
// init: zero h ping-pong buffers + barrier state
// ---------------------------------------------------------------------------
__global__ void k_init(float* h1, float* h2, unsigned* bar) {
    int i = blockIdx.x * 256 + threadIdx.x;
    if (i < 2 * B_ * U_) { h1[i] = 0.0f; h2[i] = 0.0f; }
    if (i < 8) bar[i] = 0u;
}

// ---------------------------------------------------------------------------
// pack fcW [1024][32000] f32 -> bf16 B-fragment layout:
//   out[((nt*32 + kg)*32 + lane)*16 + j] = fcW[kg*32 + 16*(lane/16) + j][nt*16 + lane%16]
// so each lane's v16bf is one contiguous 32B load, elements j = K = 16*half + j.
// ---------------------------------------------------------------------------
__global__ void k_pack_fcw(const float* __restrict__ fcW, __bf16* __restrict__ out) {
    size_t id = (size_t)blockIdx.x * 256 + threadIdx.x;   // (nt, kg, lane)
    if (id >= (size_t)(V_ / 16) * 32 * 32) return;
    int    lane = (int)(id & 31);
    size_t kg   = (id >> 5) & 31;
    size_t nt   = id >> 10;
    int    col   = (int)(nt * 16 + (lane & 15));
    int    kbase = (int)(kg * 32 + 16 * (lane >> 4));
    __bf16* o = out + id * 16;
#pragma unroll
    for (int j = 0; j < 16; ++j)
        o[j] = (__bf16)fcW[(size_t)(kbase + j) * V_ + col];
}

// ---------------------------------------------------------------------------
// One 16x16 tile of (A[16xU] @ B[Ux*]), K split over 8 waves (32-deep WMMA
// chains), partials reduced through LDS. Returns this thread's (row,col)
// element of the tile.  Bcol points at &B[0][n0], ldb = row stride of B.
// ---------------------------------------------------------------------------
__device__ __forceinline__ float tile_gemm_ksplit(
        const float* __restrict__ A, const float* __restrict__ Bcol,
        int ldb, float* red) {
    const int tid  = threadIdx.x;
    const int wave = tid >> 5, lane = tid & 31;
    const int half = (lane >> 4), lr = lane & 15;
    v8f acc = {0.f, 0.f, 0.f, 0.f, 0.f, 0.f, 0.f, 0.f};
    const int kbeg = wave * (U_ / 8);
    for (int k0 = kbeg; k0 < kbeg + (U_ / 8); k0 += 4) {
        const int ka = k0 + 2 * half;
        v2f a, b;
        a.x = A[lr * U_ + ka];
        a.y = A[lr * U_ + ka + 1];
        b.x = Bcol[(size_t)ka * ldb + lr];
        b.y = Bcol[(size_t)(ka + 1) * ldb + lr];
        acc = wmma_f32(a, b, acc);
    }
#pragma unroll
    for (int r = 0; r < 8; ++r) red[wave * 256 + r * 32 + lane] = acc[r];
    __syncthreads();
    float s = 0.f;
#pragma unroll
    for (int w = 0; w < 8; ++w) s += red[w * 256 + tid];
    __syncthreads();
    return s;
}

// ---------------------------------------------------------------------------
// Persistent two-layer GRU scan. 64 blocks (one per 16-col h tile) x 256 thr.
// Ping-pong h buffers in ws; grid barrier between the two layer phases.
// Writes rnn_out[:,t,:] each step, and final h1/h2 to d_out at the end.
// ---------------------------------------------------------------------------
__global__ __launch_bounds__(256) void k_gru_scan(
        const int*   __restrict__ tok,
        const float* __restrict__ W1x, const float* __restrict__ W1h,
        const float* __restrict__ b1,
        const float* __restrict__ W2x, const float* __restrict__ W2h,
        const float* __restrict__ b2,
        float* __restrict__ h1buf, float* __restrict__ h2buf,
        float* __restrict__ rnn, unsigned* __restrict__ bar,
        float* __restrict__ out_h1, float* __restrict__ out_h2) {
    __shared__ float red[8 * 256];
    __shared__ float gh[3][256];
    __shared__ float gx[3][256];

    const int tid = threadIdx.x;
    const int row = (tid >> 5) + 8 * ((tid >> 4) & 1);  // batch index 0..15
    const int col = tid & 15;
    const int c0  = blockIdx.x * 16;                    // owned h columns
    const int u   = c0 + col;

    float* h1c = h1buf;               float* h1n = h1buf + B_ * U_;
    float* h2c = h2buf;               float* h2n = h2buf + B_ * U_;
    unsigned* cnt = bar + 1;
    unsigned  gen = 0;

    for (int t = 0; t < TD_; ++t) {
        // -------- layer 1: hproj = h1 @ W1h (+b1_rec), gates, h1' ----------
        for (int g = 0; g < 3; ++g)
            gh[g][tid] = tile_gemm_ksplit(h1c, W1h + g * U_ + c0, 3 * U_, red);
        {
            const int   tk = tok[row * TD_ + t];
            const float xz = W1x[(size_t)tk * 3 * U_ + u]           + b1[u];
            const float xr = W1x[(size_t)tk * 3 * U_ + U_ + u]      + b1[U_ + u];
            const float xh = W1x[(size_t)tk * 3 * U_ + 2 * U_ + u]  + b1[2 * U_ + u];
            const float hz = gh[0][tid] + b1[3 * U_ + u];
            const float hr = gh[1][tid] + b1[3 * U_ + U_ + u];
            const float hh = gh[2][tid] + b1[3 * U_ + 2 * U_ + u];
            const float hp = h1c[row * U_ + u];
            const float z  = sigm(xz + hz);
            const float r  = sigm(xr + hr);
            const float hc = tanhf(xh + r * hh);
            h1n[row * U_ + u] = z * hp + (1.0f - z) * hc;
        }
        gsync(bar, cnt, gen++, gridDim.x);

        // -------- layer 2: x2 = h1' @ W2x, hproj = h2 @ W2h, gates ---------
        for (int g = 0; g < 3; ++g)
            gx[g][tid] = tile_gemm_ksplit(h1n, W2x + g * U_ + c0, 3 * U_, red);
        for (int g = 0; g < 3; ++g)
            gh[g][tid] = tile_gemm_ksplit(h2c, W2h + g * U_ + c0, 3 * U_, red);
        {
            const float xz = gx[0][tid] + b2[u];
            const float xr = gx[1][tid] + b2[U_ + u];
            const float xh = gx[2][tid] + b2[2 * U_ + u];
            const float hz = gh[0][tid] + b2[3 * U_ + u];
            const float hr = gh[1][tid] + b2[3 * U_ + U_ + u];
            const float hh = gh[2][tid] + b2[3 * U_ + 2 * U_ + u];
            const float hp = h2c[row * U_ + u];
            const float z  = sigm(xz + hz);
            const float r  = sigm(xr + hr);
            const float hc = tanhf(xh + r * hh);
            const float hn = z * hp + (1.0f - z) * hc;
            h2n[row * U_ + u] = hn;
            rnn[((size_t)row * TD_ + t) * U_ + u] = hn;
        }
        gsync(bar, cnt, gen++, gridDim.x);

        { float* s = h1c; h1c = h1n; h1n = s; }
        { float* s = h2c; h2c = h2n; h2n = s; }
    }
    out_h1[row * U_ + u] = h1c[row * U_ + u];
    out_h2[row * U_ + u] = h2c[row * U_ + u];
}

// ---------------------------------------------------------------------------
// ep[b] = enc[b] @ Wa^T  (folds attention W1 into the encoder side once)
// grid: (b, m-tile) = 16*8 blocks x 8 waves; each wave covers 8 n-tiles.
// ---------------------------------------------------------------------------
__global__ __launch_bounds__(256) void k_enc_proj(
        const float* __restrict__ enc, const float* __restrict__ Wa,
        float* __restrict__ ep) {
    const int tid  = threadIdx.x;
    const int wave = tid >> 5, lane = tid & 31;
    const int half = lane >> 4, lr = lane & 15;
    const int b  = blockIdx.x >> 3;
    const int mt = blockIdx.x & 7;
    const float* A = enc + ((size_t)b * TE_ + mt * 16) * U_;
    float*       D = ep  + ((size_t)b * TE_ + mt * 16) * U_;
    for (int nt = wave; nt < U_ / 16; nt += 8) {
        const int n0 = nt * 16;
        v8f acc = {0.f, 0.f, 0.f, 0.f, 0.f, 0.f, 0.f, 0.f};
        for (int k0 = 0; k0 < U_; k0 += 4) {
            const int ka = k0 + 2 * half;
            v2f a, bb;
            a.x  = A[lr * U_ + ka];
            a.y  = A[lr * U_ + ka + 1];
            bb.x = Wa[(size_t)(n0 + lr) * U_ + ka];       // B[k][n] = Wa^T
            bb.y = Wa[(size_t)(n0 + lr) * U_ + ka + 1];
            acc = wmma_f32(a, bb, acc);
        }
#pragma unroll
        for (int r = 0; r < 8; ++r)
            D[(r + 8 * half) * U_ + n0 + lr] = acc[r];
    }
}

// ---------------------------------------------------------------------------
// Fused attention per (b, t-tile): scores = rnn @ ep^T, mask, softmax,
// attn -> d_out (NT store, write-once), context = attn @ enc (* dec_mask).
// ---------------------------------------------------------------------------
__global__ __launch_bounds__(256) void k_attn(
        const float* __restrict__ rnn, const float* __restrict__ ep,
        const float* __restrict__ enc,
        const int* __restrict__ enc_mask, const int* __restrict__ dec_mask,
        float* __restrict__ ctx, float* __restrict__ attn_out) {
    __shared__ float sc[16][TE_ + 4];
    const int tid  = threadIdx.x;
    const int wave = tid >> 5, lane = tid & 31;
    const int half = lane >> 4, lr = lane & 15;
    const int b  = blockIdx.x >> 3;
    const int t0 = (blockIdx.x & 7) * 16;
    const float* A = rnn + ((size_t)b * TD_ + t0) * U_;

    // scores tile: each wave owns one of the 8 N-tiles (TE=128)
    {
        const int n0 = wave * 16;
        const float* Bp = ep + (size_t)b * TE_ * U_;      // B[u][s] = ep[s][u]
        v8f acc = {0.f, 0.f, 0.f, 0.f, 0.f, 0.f, 0.f, 0.f};
        for (int k0 = 0; k0 < U_; k0 += 4) {
            const int ka = k0 + 2 * half;
            v2f a, bb;
            a.x  = A[lr * U_ + ka];
            a.y  = A[lr * U_ + ka + 1];
            bb.x = Bp[(size_t)(n0 + lr) * U_ + ka];
            bb.y = Bp[(size_t)(n0 + lr) * U_ + ka + 1];
            acc = wmma_f32(a, bb, acc);
        }
#pragma unroll
        for (int r = 0; r < 8; ++r) {
            const int rr = r + 8 * half, s = n0 + lr;
            sc[rr][s] = (enc_mask[b * TE_ + s] != 0) ? acc[r] : NEG_;
        }
    }
    __syncthreads();

    if (tid < 16) {                               // softmax per query row
        float mx = -3.4e38f;
        for (int s = 0; s < TE_; ++s) mx = fmaxf(mx, sc[tid][s]);
        float sum = 0.f;
        for (int s = 0; s < TE_; ++s) { float e = __expf(sc[tid][s] - mx); sc[tid][s] = e; sum += e; }
        const float inv = 1.0f / sum;
        for (int s = 0; s < TE_; ++s) sc[tid][s] *= inv;
    }
    __syncthreads();

    for (int i = tid; i < 16 * TE_; i += 256) {   // emit attn (pre dec-mask)
        const int rr = i >> 7, s = i & (TE_ - 1);
        __builtin_nontemporal_store(sc[rr][s],
            &attn_out[((size_t)b * TD_ + t0 + rr) * TE_ + s]);
    }

    // context = attn @ enc[b], masked by dec_mask per query row
    const float* E = enc + (size_t)b * TE_ * U_;
    for (int nt = wave; nt < U_ / 16; nt += 8) {
        const int n0 = nt * 16;
        v8f acc = {0.f, 0.f, 0.f, 0.f, 0.f, 0.f, 0.f, 0.f};
        for (int k0 = 0; k0 < TE_; k0 += 4) {
            const int ka = k0 + 2 * half;
            v2f a, bb;
            a.x  = sc[lr][ka];
            a.y  = sc[lr][ka + 1];
            bb.x = E[(size_t)ka * U_ + n0 + lr];
            bb.y = E[(size_t)(ka + 1) * U_ + n0 + lr];
            acc = wmma_f32(a, bb, acc);
        }
#pragma unroll
        for (int r = 0; r < 8; ++r) {
            const int rr = r + 8 * half;
            const float dm = (dec_mask[b * TD_ + t0 + rr] != 0) ? 1.0f : 0.0f;
            ctx[((size_t)b * TD_ + t0 + rr) * U_ + n0 + lr] = acc[r] * dm;
        }
    }
}

// ---------------------------------------------------------------------------
// last = tanh([context | rnn] @ Wc), emitted directly as bf16 for the
// logits GEMM. M=2048 (16 groups x 8 waves), N=1024 (64 tiles), K=2048.
// ---------------------------------------------------------------------------
__global__ __launch_bounds__(256) void k_wc(
        const float* __restrict__ ctx, const float* __restrict__ rnn,
        const float* __restrict__ Wc, __bf16* __restrict__ last_bf) {
    const int tid  = threadIdx.x;
    const int wave = tid >> 5, lane = tid & 31;
    const int half = lane >> 4, lr = lane & 15;
    const int mg = blockIdx.x >> 6;
    const int nt = blockIdx.x & 63;
    const int m0 = (mg * 8 + wave) * 16;
    const int n0 = nt * 16;
    const float* A1 = ctx + (size_t)m0 * U_;
    const float* A2 = rnn + (size_t)m0 * U_;
    v8f acc = {0.f, 0.f, 0.f, 0.f, 0.f, 0.f, 0.f, 0.f};
    for (int k0 = 0; k0 < 2 * U_; k0 += 4) {
        const int ka = k0 + 2 * half;          // ka,ka+1 never straddle U_
        v2f a, bb;
        if (ka < U_) { a.x = A1[lr * U_ + ka];        a.y = A1[lr * U_ + ka + 1]; }
        else         { a.x = A2[lr * U_ + ka - U_];   a.y = A2[lr * U_ + ka - U_ + 1]; }
        bb.x = Wc[(size_t)ka * U_ + n0 + lr];
        bb.y = Wc[(size_t)(ka + 1) * U_ + n0 + lr];
        acc = wmma_f32(a, bb, acc);
    }
#pragma unroll
    for (int r = 0; r < 8; ++r)
        last_bf[(size_t)(m0 + r + 8 * half) * U_ + n0 + lr] = (__bf16)tanhf(acc[r]);
}

// ---------------------------------------------------------------------------
// logits = last_bf16 @ fcW_bf16 + fcb  (bf16 WMMA 16x16x32, f32 accumulate).
// grid: (nt=2000, mg=16); 8 waves share each packed B fragment across 8
// M-tiles. Output is written with non-temporal stores: 262MB write-once
// stream must not evict the L2-resident packed fcW between M-groups.
// ---------------------------------------------------------------------------
__global__ __launch_bounds__(256) void k_logits(
        const __bf16* __restrict__ last_bf, const __bf16* __restrict__ fcWb,
        const float* __restrict__ fcb, float* __restrict__ out) {
    const int tid  = threadIdx.x;
    const int wave = tid >> 5, lane = tid & 31;
    const int half = lane >> 4, lr = lane & 15;
    const int nt = blockIdx.x;                  // 0..1999
    const int mg = blockIdx.y;                  // 0..15
    const int m0 = (mg * 8 + wave) * 16;
    const int n0 = nt * 16;
    const __bf16* A  = last_bf + (size_t)m0 * U_;
    const __bf16* Bp = fcWb + (size_t)nt * 32 * 32 * 16;
    v8f acc = {0.f, 0.f, 0.f, 0.f, 0.f, 0.f, 0.f, 0.f};
    for (int kg = 0; kg < U_ / 32; ++kg) {
        const int k0 = kg * 32;
        const v16bf av = *(const v16bf*)(A + (size_t)lr * U_ + k0 + 16 * half);
        const v16bf bv = *(const v16bf*)(Bp + ((size_t)kg * 32 + lane) * 16);
        acc = wmma_bf16(av, bv, acc);
    }
    const float bias = fcb[n0 + lr];
#pragma unroll
    for (int r = 0; r < 8; ++r) {
        const size_t m = (size_t)m0 + r + 8 * half;
        __builtin_nontemporal_store(acc[r] + bias, &out[m * V_ + n0 + lr]);
    }
}

// ---------------------------------------------------------------------------
extern "C" void kernel_launch(void* const* d_in, const int* in_sizes, int n_in,
                              void* d_out, int out_size, void* d_ws, size_t ws_size,
                              hipStream_t stream) {
    (void)in_sizes; (void)n_in; (void)out_size; (void)ws_size;
    const int*   tok      = (const int*)  d_in[0];
    const int*   dec_mask = (const int*)  d_in[1];
    const float* enc      = (const float*)d_in[2];
    const int*   enc_mask = (const int*)  d_in[3];
    const float* W1x      = (const float*)d_in[4];
    const float* W1h      = (const float*)d_in[5];
    const float* b1       = (const float*)d_in[6];
    const float* W2x      = (const float*)d_in[7];
    const float* W2h      = (const float*)d_in[8];
    const float* b2       = (const float*)d_in[9];
    const float* Wa       = (const float*)d_in[10];
    const float* Wc       = (const float*)d_in[11];
    const float* fcW      = (const float*)d_in[12];
    const float* fcb      = (const float*)d_in[13];

    float* logits = (float*)d_out;                            // [16,128,32000]
    float* attn   = logits + (size_t)B_ * TD_ * V_;           // [16,128,128]
    float* oh1    = attn   + (size_t)B_ * TD_ * TE_;          // [16,1024]
    float* oh2    = oh1    + (size_t)B_ * U_;                 // [16,1024]

    float* w = (float*)d_ws;
    unsigned* bar = (unsigned*)w;                 w += 64;
    float* h1   = w;                              w += 2 * B_ * U_;
    float* h2   = w;                              w += 2 * B_ * U_;
    float* rnn  = w;                              w += (size_t)B_ * TD_ * U_;
    float* ep   = w;                              w += (size_t)B_ * TE_ * U_;
    float* ctx  = w;                              w += (size_t)B_ * TD_ * U_;
    __bf16* lastbf = (__bf16*)w;                  w += ((size_t)B_ * TD_ * U_) / 2;
    __bf16* fcwb   = (__bf16*)w;                  /* 1024*32000 bf16 */

    k_init    <<<128, 256, 0, stream>>>(h1, h2, bar);
    k_pack_fcw<<<8000, 256, 0, stream>>>(fcW, fcwb);
    k_gru_scan<<<64, 256, 0, stream>>>(tok, W1x, W1h, b1, W2x, W2h, b2,
                                       h1, h2, rnn, bar, oh1, oh2);
    k_enc_proj<<<B_ * (TE_ / 16), 256, 0, stream>>>(enc, Wa, ep);
    k_attn    <<<B_ * (TD_ / 16), 256, 0, stream>>>(rnn, ep, enc, enc_mask,
                                                    dec_mask, ctx, attn);
    k_wc      <<<16 * 64, 256, 0, stream>>>(ctx, rnn, Wc, lastbf);
    k_logits  <<<dim3(V_ / 16, 16), 256, 0, stream>>>(lastbf, fcwb, fcb, logits);
}
